// SelfAttention_25288767439125
// MI455X (gfx1250) — compile-verified, matrix operand order checked by reading, not run on hardware
//
#include <hip/hip_runtime.h>
#include <hip/hip_bf16.h>

typedef _Float16 h16;
typedef __attribute__((ext_vector_type(16))) _Float16 v16h;
typedef __attribute__((ext_vector_type(8)))  _Float16 v8h;
typedef __attribute__((ext_vector_type(8)))  float    v8f;
typedef __attribute__((ext_vector_type(4)))  int      v4i;

#define B_  4
#define C_  1024
#define T_  2048
#define H_  16
#define D_  64
#define BT_ (B_ * T_)

#ifndef __has_builtin
#define __has_builtin(x) 0
#endif

#if __has_builtin(__builtin_amdgcn_global_load_async_to_lds_b128) && \
    __has_builtin(__builtin_amdgcn_s_wait_asynccnt)
#define ASYNC_LDS 1
#else
#define ASYNC_LDS 0
#endif

#if ASYNC_LDS
// Signature (from hipcc diagnostic / LLVM decl "vV4i*1V4i*3IiIi"):
//   (global v4i*, LDS v4i*, imm offset, imm cpol)
__device__ __forceinline__ void async_cp16B(h16* l, const h16* g) {
  __builtin_amdgcn_global_load_async_to_lds_b128(
      (__attribute__((address_space(1))) v4i*)(g),
      (__attribute__((address_space(3))) v4i*)(l), 0, 0);
}
#endif

// ---------------------------------------------------------------------------
// WMMA fragment loader: 16x32 f16 fragment from a K-contiguous matrix.
// ISA 16-bit A layout: lane L (<16) holds row M=L, K={0..7,16..23}; lane L+16
// holds row M=L, K={8..15,24..31}.  Two 16B chunks per lane.  B fragments use
// the mirrored layout (col = lane&15), so B operands are stored K-contiguous.
// ---------------------------------------------------------------------------
__device__ __forceinline__ v16h load_frag(const h16* __restrict__ base, int ld) {
  int lane = threadIdx.x & 31;
  int r  = lane & 15;
  int b0 = (lane >> 4) << 3;
  const h16* p = base + r * ld + b0;
  v8h lo = *(const v8h*)(p);
  v8h hi = *(const v8h*)(p + 16);
  return __builtin_shufflevector(lo, hi, 0,1,2,3,4,5,6,7,8,9,10,11,12,13,14,15);
}

__device__ __forceinline__ v8f wmma_f16(v16h a, v16h b, v8f c) {
  return __builtin_amdgcn_wmma_f32_16x16x32_f16(false, a, false, b, (short)0, c,
                                                false, false);
}

__device__ __forceinline__ float redmax16(float x) {
  for (int m = 1; m < 16; m <<= 1) x = fmaxf(x, __shfl_xor(x, m, 16));
  return x;
}
__device__ __forceinline__ float redsum16(float x) {
  for (int m = 1; m < 16; m <<= 1) x += __shfl_xor(x, m, 16);
  return x;
}

// ---------------------------------------------------------------------------
// Shared GEMM main loop: 128x128 block tile, BK=32, 8 waves (2x4), each wave
// 64x32 (4x2 WMMA tiles).  Double-buffered LDS; ASYNCcnt global->LDS copies
// when the toolchain exposes them, otherwise a register-staged software
// pipeline (prefetch next tile after the barrier, compute current).
// One barrier per K-step: __syncthreads' implicit s_wait_dscnt makes the
// alternate buffer WAR-safe.
// ---------------------------------------------------------------------------
#define LDS_TILE 5120  // 128 rows * 40 halves (8-half pad per row)

__device__ __forceinline__ void gemm_mainloop(
    const h16* __restrict__ Ag, const h16* __restrict__ Bg, int m0, int n0,
    h16* As, h16* Bs, v8f acc[4][2]) {
  int tid = threadIdx.x;
  int wid = tid >> 5;
  int wm = wid >> 2;  // 0..1
  int wn = wid & 3;   // 0..3
  const int NT = C_ / 32;

#if ASYNC_LDS
  auto issue = [&](int kt, int sel) {
    int k0 = kt * 32;
    h16* as = As + sel * LDS_TILE;
    h16* bs = Bs + sel * LDS_TILE;
    for (int i = 0; i < 2; ++i) {
      int ci = tid + i * 256;
      int row = ci >> 2;
      int kc  = ci & 3;
      async_cp16B(as + row * 40 + kc * 8,
                  Ag + (size_t)(m0 + row) * C_ + k0 + kc * 8);
      async_cp16B(bs + row * 40 + kc * 8,
                  Bg + (size_t)(n0 + row) * C_ + k0 + kc * 8);
    }
  };
  issue(0, 0);
  for (int kt = 0; kt < NT; ++kt) {
    __builtin_amdgcn_s_wait_asynccnt(0);
    __syncthreads();
    if (kt + 1 < NT) issue(kt + 1, (kt + 1) & 1);
    const h16* as = As + (kt & 1) * LDS_TILE;
    const h16* bs = Bs + (kt & 1) * LDS_TILE;
    v16h af[4], bf[2];
    for (int mi = 0; mi < 4; ++mi)
      af[mi] = load_frag(as + (wm * 64 + mi * 16) * 40, 40);
    for (int ni = 0; ni < 2; ++ni)
      bf[ni] = load_frag(bs + (wn * 32 + ni * 16) * 40, 40);
    for (int mi = 0; mi < 4; ++mi)
      for (int ni = 0; ni < 2; ++ni)
        acc[mi][ni] = wmma_f16(af[mi], bf[ni], acc[mi][ni]);
  }
#else
  v8h ra[2], rb[2];
  auto loadr = [&](int kt) {
    int k0 = kt * 32;
    for (int i = 0; i < 2; ++i) {
      int ci = tid + i * 256;
      int row = ci >> 2;
      int kc  = ci & 3;
      ra[i] = *(const v8h*)(Ag + (size_t)(m0 + row) * C_ + k0 + kc * 8);
      rb[i] = *(const v8h*)(Bg + (size_t)(n0 + row) * C_ + k0 + kc * 8);
    }
  };
  loadr(0);
  for (int kt = 0; kt < NT; ++kt) {
    h16* asw = As + (kt & 1) * LDS_TILE;
    h16* bsw = Bs + (kt & 1) * LDS_TILE;
    for (int i = 0; i < 2; ++i) {
      int ci = tid + i * 256;
      int row = ci >> 2;
      int kc  = ci & 3;
      *(v8h*)(asw + row * 40 + kc * 8) = ra[i];
      *(v8h*)(bsw + row * 40 + kc * 8) = rb[i];
    }
    __syncthreads();
    if (kt + 1 < NT) loadr(kt + 1);  // overlap with compute below
    v16h af[4], bf[2];
    for (int mi = 0; mi < 4; ++mi)
      af[mi] = load_frag(asw + (wm * 64 + mi * 16) * 40, 40);
    for (int ni = 0; ni < 2; ++ni)
      bf[ni] = load_frag(bsw + (wn * 32 + ni * 16) * 40, 40);
    for (int mi = 0; mi < 4; ++mi)
      for (int ni = 0; ni < 2; ++ni)
        acc[mi][ni] = wmma_f16(af[mi], bf[ni], acc[mi][ni]);
  }
#endif
}

// ---------------------------------------------------------------------------
// K1: LayerNorm over channels + transpose to [B*T, C] f16.
// ---------------------------------------------------------------------------
__global__ __launch_bounds__(256) void sa_ln_kernel(
    const float* __restrict__ x, const float* __restrict__ gamma,
    const float* __restrict__ beta, h16* __restrict__ xn) {
  __shared__ float redS[32][8];
  __shared__ float redQ[32][8];
  __shared__ float smean[32];
  __shared__ float srstd[32];
  int b  = blockIdx.x;
  int t0 = blockIdx.y * 32;
  int tid = threadIdx.x;
  int tt = tid & 31;
  int cc = tid >> 5;  // 0..7
  const float* xp = x + (size_t)b * C_ * T_ + t0 + tt;
  float s = 0.f, ss = 0.f;
  for (int c = cc * 128; c < cc * 128 + 128; ++c) {
    float v = xp[(size_t)c * T_];
    s += v; ss += v * v;
  }
  redS[tt][cc] = s; redQ[tt][cc] = ss;
  __syncthreads();
  if (cc == 0) {
    float a = 0.f, q = 0.f;
    for (int j = 0; j < 8; ++j) { a += redS[tt][j]; q += redQ[tt][j]; }
    float mu  = a * (1.0f / C_);
    float var = q * (1.0f / C_) - mu * mu;
    smean[tt] = mu;
    srstd[tt] = rsqrtf(var + 1e-5f);
  }
  __syncthreads();
  float mu = smean[tt], rs = srstd[tt];
  h16* xrow = xn + (size_t)(b * T_ + t0 + tt) * C_;
  for (int c0 = 0; c0 < C_; c0 += 8) {
    int c = c0 + cc;
    float v = xp[(size_t)c * T_];
    xrow[c] = (h16)((v - mu) * rs * gamma[c] + beta[c]);
  }
}

// ---------------------------------------------------------------------------
// K2: weight transpose + f32->f16:  Wt[n][k] = (f16) W[k][n].
// ---------------------------------------------------------------------------
__global__ __launch_bounds__(256) void sa_wtrans_kernel(
    const float* __restrict__ w, h16* __restrict__ wt, int K, int N) {
  __shared__ float tile[32][33];
  int n0 = blockIdx.x * 32;
  int k0 = blockIdx.y * 32;
  int tid = threadIdx.x;
  int r = tid >> 5;
  int c = tid & 31;
  for (int i = 0; i < 4; ++i)
    tile[r + i * 8][c] = w[(size_t)(k0 + r + i * 8) * N + n0 + c];
  __syncthreads();
  for (int i = 0; i < 4; ++i)
    wt[(size_t)(n0 + r + i * 8) * K + k0 + c] = (h16)tile[c][r + i * 8];
}

// ---------------------------------------------------------------------------
// K3: QKV GEMM, epilogue scatters to Q [B,H,T,D] (pre-scaled by 1/sqrt(D)),
// K [B,H,T,D], V transposed [B,H,D,T].  s = q/k/v select is uniform per block.
// ---------------------------------------------------------------------------
__global__ __launch_bounds__(256) void sa_qkv_gemm_kernel(
    const h16* __restrict__ xn, const h16* __restrict__ wt,
    const float* __restrict__ bias, h16* __restrict__ qb,
    h16* __restrict__ kb, h16* __restrict__ vtb) {
  __shared__ h16 As[2 * LDS_TILE];
  __shared__ h16 Bs[2 * LDS_TILE];
  int tid = threadIdx.x;
  int m0 = blockIdx.x * 128;
  int n0 = blockIdx.y * 128;
  v8f acc[4][2];
  for (int i = 0; i < 4; ++i)
    for (int j = 0; j < 2; ++j)
      acc[i][j] = (v8f){0.f,0.f,0.f,0.f,0.f,0.f,0.f,0.f};

  gemm_mainloop(xn, wt, m0, n0, As, Bs, acc);

  int wid = tid >> 5;
  int wm = wid >> 2;
  int wn = wid & 3;
  int lane = tid & 31;
  int hi = lane >> 4;
  int nc = lane & 15;
  const int s = n0 >> 10;  // uniform: 1024 % 128 == 0
  const float scaleQ = 0.125f;

  if (s == 0) {
    for (int mi = 0; mi < 4; ++mi)
      for (int ni = 0; ni < 2; ++ni) {
        int n = n0 + wn * 32 + ni * 16 + nc;
        int c = n & (C_ - 1);
        int bh0 = c >> 6;
        int dd = c & (D_ - 1);
        float bv = bias[n];
        for (int v = 0; v < 8; ++v) {
          int m = m0 + wm * 64 + mi * 16 + v + hi * 8;
          int bh = (m >> 11) * H_ + bh0;
          int t  = m & (T_ - 1);
          qb[((size_t)bh * T_ + t) * D_ + dd] =
              (h16)((acc[mi][ni][v] + bv) * scaleQ);
        }
      }
  } else if (s == 1) {
    for (int mi = 0; mi < 4; ++mi)
      for (int ni = 0; ni < 2; ++ni) {
        int n = n0 + wn * 32 + ni * 16 + nc;
        int c = n & (C_ - 1);
        int bh0 = c >> 6;
        int dd = c & (D_ - 1);
        float bv = bias[n];
        for (int v = 0; v < 8; ++v) {
          int m = m0 + wm * 64 + mi * 16 + v + hi * 8;
          int bh = (m >> 11) * H_ + bh0;
          int t  = m & (T_ - 1);
          kb[((size_t)bh * T_ + t) * D_ + dd] = (h16)(acc[mi][ni][v] + bv);
        }
      }
  } else {
    for (int mi = 0; mi < 4; ++mi)
      for (int ni = 0; ni < 2; ++ni) {
        int n = n0 + wn * 32 + ni * 16 + nc;
        int c = n & (C_ - 1);
        int bh0 = c >> 6;
        int dd = c & (D_ - 1);
        float bv = bias[n];
        for (int v = 0; v < 8; ++v) {
          int m = m0 + wm * 64 + mi * 16 + v + hi * 8;
          int bh = (m >> 11) * H_ + bh0;
          int t  = m & (T_ - 1);
          vtb[((size_t)bh * D_ + dd) * T_ + t] = (h16)(acc[mi][ni][v] + bv);
        }
      }
  }
}

// ---------------------------------------------------------------------------
// K4: flash-style attention.  grid (B*H, T/128), block 256 (8 waves).
// Q fragments in registers for the whole KV loop; K/V from global (L2-resident
// on the 192MB L2); next tile prefetched; P re-laid-out C->A via per-wave
// private LDS (same-wave DS ordering, no barriers).
// ---------------------------------------------------------------------------
__global__ __launch_bounds__(256) void sa_attn_kernel(
    const h16* __restrict__ qb, const h16* __restrict__ kb,
    const h16* __restrict__ vtb, h16* __restrict__ ob) {
  __shared__ h16 plds[8 * 16 * 32];
  int tid = threadIdx.x;
  int lane = tid & 31;
  int wid = tid >> 5;
  int bh = blockIdx.x;
  int q0 = blockIdx.y * 128 + wid * 16;
  const h16* qp = qb + ((size_t)bh * T_ + q0) * D_;
  const h16* kp = kb + (size_t)bh * T_ * D_;
  const h16* vp = vtb + (size_t)bh * D_ * T_;
  h16* pl = plds + wid * 16 * 32;
  int hi = lane >> 4;
  int nc = lane & 15;

  v16h qf0 = load_frag(qp, D_);
  v16h qf1 = load_frag(qp + 32, D_);
  v8f o0 = (v8f){0.f,0.f,0.f,0.f,0.f,0.f,0.f,0.f};
  v8f o1 = o0, o2 = o0, o3 = o0;
  float rmax[8], rsum[8];
  for (int v = 0; v < 8; ++v) { rmax[v] = -1e30f; rsum[v] = 0.f; }

  for (int kv = 0; kv < T_; kv += 32) {
    const h16* kpp = kp + (size_t)kv * D_;
    if (kv + 32 < T_) {  // pull next K/V tile toward the WGP
      __builtin_prefetch(kpp + 32 * D_ + lane * D_, 0, 3);
      __builtin_prefetch(vp + (size_t)lane * T_ + kv + 32, 0, 3);
      __builtin_prefetch(vp + (size_t)(32 + lane) * T_ + kv + 32, 0, 3);
    }
    v8f s0 = (v8f){0.f,0.f,0.f,0.f,0.f,0.f,0.f,0.f};
    v8f s1 = s0;
    s0 = wmma_f16(qf0, load_frag(kpp, D_), s0);
    s0 = wmma_f16(qf1, load_frag(kpp + 32, D_), s0);
    s1 = wmma_f16(qf0, load_frag(kpp + 16 * D_, D_), s1);
    s1 = wmma_f16(qf1, load_frag(kpp + 16 * D_ + 32, D_), s1);

    float al[8];
    for (int v = 0; v < 8; ++v) {
      float tm = redmax16(fmaxf(s0[v], s1[v]));
      float nm = fmaxf(rmax[v], tm);
      float alpha = __expf(rmax[v] - nm);
      rmax[v] = nm;
      float p0 = __expf(s0[v] - nm);
      float p1 = __expf(s1[v] - nm);
      rsum[v] = rsum[v] * alpha + redsum16(p0 + p1);
      al[v] = alpha;
      pl[(v + hi * 8) * 32 + nc]      = (h16)p0;
      pl[(v + hi * 8) * 32 + 16 + nc] = (h16)p1;
    }
    for (int v = 0; v < 8; ++v) {
      o0[v] *= al[v]; o1[v] *= al[v]; o2[v] *= al[v]; o3[v] *= al[v];
    }
    v16h pa = load_frag(pl, 32);  // same-wave LDS ordering vs stores above
    o0 = wmma_f16(pa, load_frag(vp + 0 * 16 * T_ + kv, T_), o0);
    o1 = wmma_f16(pa, load_frag(vp + 1 * 16 * T_ + kv, T_), o1);
    o2 = wmma_f16(pa, load_frag(vp + 2 * 16 * T_ + kv, T_), o2);
    o3 = wmma_f16(pa, load_frag(vp + 3 * 16 * T_ + kv, T_), o3);
  }

  int bb = bh >> 4;
  int hh = bh & (H_ - 1);
  for (int v = 0; v < 8; ++v) {
    float inv = 1.0f / rsum[v];
    int qrow = q0 + v + hi * 8;
    h16* orow = ob + ((size_t)(bb * T_ + qrow)) * C_ + hh * D_;
    orow[0  + nc] = (h16)(o0[v] * inv);
    orow[16 + nc] = (h16)(o1[v] * inv);
    orow[32 + nc] = (h16)(o2[v] * inv);
    orow[48 + nc] = (h16)(o3[v] * inv);
  }
}

// ---------------------------------------------------------------------------
// K5: output projection + bias + residual, write y as [B, C, T] f32.
// ---------------------------------------------------------------------------
__global__ __launch_bounds__(256) void sa_proj_gemm_kernel(
    const h16* __restrict__ ob, const h16* __restrict__ wt,
    const float* __restrict__ bias, const float* __restrict__ x,
    float* __restrict__ y) {
  __shared__ h16 As[2 * LDS_TILE];
  __shared__ h16 Bs[2 * LDS_TILE];
  int tid = threadIdx.x;
  int m0 = blockIdx.x * 128;
  int n0 = blockIdx.y * 128;
  v8f acc[4][2];
  for (int i = 0; i < 4; ++i)
    for (int j = 0; j < 2; ++j)
      acc[i][j] = (v8f){0.f,0.f,0.f,0.f,0.f,0.f,0.f,0.f};

  gemm_mainloop(ob, wt, m0, n0, As, Bs, acc);

  int wid = tid >> 5;
  int wm = wid >> 2;
  int wn = wid & 3;
  int lane = tid & 31;
  int hi = lane >> 4;
  int nc = lane & 15;
  for (int mi = 0; mi < 4; ++mi) {
    for (int ni = 0; ni < 2; ++ni) {
      int n = n0 + wn * 32 + ni * 16 + nc;
      float bv = bias[n];
      for (int v = 0; v < 8; ++v) {
        int m = m0 + wm * 64 + mi * 16 + v + hi * 8;
        int bb = m >> 11;
        int t  = m & (T_ - 1);
        size_t idx = ((size_t)bb * C_ + n) * T_ + t;
        y[idx] = x[idx] + acc[mi][ni][v] + bv;
      }
    }
  }
}

// ---------------------------------------------------------------------------
extern "C" void kernel_launch(void* const* d_in, const int* in_sizes, int n_in,
                              void* d_out, int out_size, void* d_ws,
                              size_t ws_size, hipStream_t stream) {
  const float* x     = (const float*)d_in[0];
  const float* Wqkv  = (const float*)d_in[1];
  const float* bqkv  = (const float*)d_in[2];
  const float* Wproj = (const float*)d_in[3];
  const float* bproj = (const float*)d_in[4];
  const float* gamma = (const float*)d_in[5];
  const float* beta  = (const float*)d_in[6];
  float* y = (float*)d_out;

  char* ws = (char*)d_ws;
  size_t off = 0;
  h16* xn  = (h16*)(ws + off); off += (size_t)BT_ * C_ * 2;       // 16 MB
  h16* wtq = (h16*)(ws + off); off += (size_t)3 * C_ * C_ * 2;    //  6 MB
  h16* wtp = (h16*)(ws + off); off += (size_t)C_ * C_ * 2;        //  2 MB
  h16* qb  = (h16*)(ws + off); off += (size_t)BT_ * C_ * 2;       // 16 MB
  h16* kb  = (h16*)(ws + off); off += (size_t)BT_ * C_ * 2;       // 16 MB
  h16* vtb = (h16*)(ws + off); off += (size_t)BT_ * C_ * 2;       // 16 MB
  h16* ob  = (h16*)(ws + off); off += (size_t)BT_ * C_ * 2;       // 16 MB

  sa_ln_kernel<<<dim3(B_, T_ / 32), 256, 0, stream>>>(x, gamma, beta, xn);
  sa_wtrans_kernel<<<dim3(3 * C_ / 32, C_ / 32), 256, 0, stream>>>(Wqkv, wtq,
                                                                   C_, 3 * C_);
  sa_wtrans_kernel<<<dim3(C_ / 32, C_ / 32), 256, 0, stream>>>(Wproj, wtp, C_,
                                                               C_);
  sa_qkv_gemm_kernel<<<dim3(BT_ / 128, 3 * C_ / 128), 256, 0, stream>>>(
      xn, wtq, bqkv, qb, kb, vtb);
  sa_attn_kernel<<<dim3(B_ * H_, T_ / 128), 256, 0, stream>>>(qb, kb, vtb, ob);
  sa_proj_gemm_kernel<<<dim3(BT_ / 128, C_ / 128), 256, 0, stream>>>(
      ob, wtp, bproj, x, y);
}
